// Attention_20203526160915
// MI455X (gfx1250) — 16-bit and 8-bit matrix operand order repaired and checked by reading, compiled, not run on hardware
//
#include <hip/hip_runtime.h>
#include <hip/hip_bf16.h>
#include <math.h>

typedef __attribute__((ext_vector_type(8)))  int      v8i;
typedef __attribute__((ext_vector_type(8)))  float    v8f;
typedef __attribute__((ext_vector_type(16))) _Float16 v16h;

#define B_   64
#define N_   197
#define C_   768
#define H_   12
#define HD   64
#define M3   2304
#define BN   12608
#define NP   208
#define NP2  224
#define NT   13

struct Scalars {
  unsigned mm[5][2];
  float scale[5];
  float zp[5];
};

__device__ __forceinline__ unsigned f_enc(float f){
  unsigned u = __float_as_uint(f);
  return (u & 0x80000000u) ? ~u : (u | 0x80000000u);
}
__device__ __forceinline__ float f_dec(unsigned u){
  return __uint_as_float((u & 0x80000000u) ? (u & 0x7fffffffu) : ~u);
}
__device__ __forceinline__ float round_ln_f(float x){
  float y = floorf(log2f(x));
  float p = exp2f(y);
  return y + (((x - p) > (2.f * p - x)) ? 1.f : 0.f);
}
__device__ __forceinline__ void wave_mm_atomic(float mn, float mx, unsigned* slot){
#pragma unroll
  for (int o = 16; o > 0; o >>= 1){
    mn = fminf(mn, __shfl_xor(mn, o, 32));
    mx = fmaxf(mx, __shfl_xor(mx, o, 32));
  }
  if ((threadIdx.x & 31) == 0){
    atomicMin(&slot[0], f_enc(mn));
    atomicMax(&slot[1], f_enc(mx));
  }
}

__global__ void k_init(Scalars* sc, unsigned* gmax, unsigned* wmax){
  int t = blockIdx.x * 256 + threadIdx.x;
  if (t < C_){ gmax[t] = 0u; wmax[t] = 0u; }
  if (t < 5){ sc->mm[t][0] = 0xFFFFFFFFu; sc->mm[t][1] = 0u; }
}

__global__ __launch_bounds__(256) void k_absmax_cols(const float* w, unsigned* omax, int rpb){
  int r0 = blockIdx.x * rpb, t = threadIdx.x;
  float m0 = 0.f, m1 = 0.f, m2 = 0.f;
  for (int r = 0; r < rpb; ++r){
    const float* row = w + (size_t)(r0 + r) * C_;
    m0 = fmaxf(m0, fabsf(row[t]));
    m1 = fmaxf(m1, fabsf(row[t + 256]));
    m2 = fmaxf(m2, fabsf(row[t + 512]));
  }
  atomicMax(&omax[t],       __float_as_uint(m0));
  atomicMax(&omax[t + 256], __float_as_uint(m1));
  atomicMax(&omax[t + 512], __float_as_uint(m2));
}

__global__ void k_cs(const float* gmax, const float* wmax, float* cs){
  int t = blockIdx.x * 256 + threadIdx.x;
  if (t < C_){
    float g = fmaxf(gmax[t], 1e-8f);
    float w = fmaxf(wmax[t], 1e-8f);
    float c = powf(g, 0.35f) / powf(w, 0.65f);
    cs[t] = exp2f(round_ln_f(c));
  }
}

__global__ __launch_bounds__(256) void k_xs_minmax(const float* x, const float* cs, Scalars* sc){
  int r0 = blockIdx.x * 64, t = threadIdx.x;
  float i0 = 1.f / cs[t], i1 = 1.f / cs[t + 256], i2 = 1.f / cs[t + 512];
  float mn = 3.4e38f, mx = -3.4e38f;
  for (int r = 0; r < 64; ++r){
    const float* row = x + (size_t)(r0 + r) * C_;
    float a = row[t] * i0, b = row[t + 256] * i1, c = row[t + 512] * i2;
    mn = fminf(mn, fminf(a, fminf(b, c)));
    mx = fmaxf(mx, fmaxf(a, fmaxf(b, c)));
  }
  wave_mm_atomic(mn, mx, sc->mm[0]);
}

__global__ void k_finalize(Scalars* sc, int slot){
  float mn = f_dec(sc->mm[slot][0]);
  float mx = f_dec(sc->mm[slot][1]);
  float s = fmaxf((mx - mn) * (1.f / 255.f), 1e-8f);
  sc->scale[slot] = s;
  sc->zp[slot] = rintf(-mn / s);
}

__global__ __launch_bounds__(256) void k_quant_w(const float* w, const float* cs,
                                                 signed char* qw, float* scw, float* rsw){
  int m = blockIdx.x, t = threadIdx.x;
  __shared__ float redm[8], reds[8];
  const float* row = w + (size_t)m * C_;
  float amax = 0.f;
  for (int c = t; c < C_; c += 256){
    float v = row[c] * (cs ? cs[c] : 1.f);
    amax = fmaxf(amax, fabsf(v));
  }
#pragma unroll
  for (int o = 16; o > 0; o >>= 1) amax = fmaxf(amax, __shfl_xor(amax, o, 32));
  if ((t & 31) == 0) redm[t >> 5] = amax;
  __syncthreads();
  if (t < 32){
    float v = (t < 8) ? redm[t] : 0.f;
#pragma unroll
    for (int o = 4; o > 0; o >>= 1) v = fmaxf(v, __shfl_xor(v, o, 32));
    if (t == 0) redm[0] = v;
  }
  __syncthreads();
  float scale = fmaxf(redm[0], 1e-8f) * (1.f / 127.f);
  float ssum = 0.f;
  for (int c = t; c < C_; c += 256){
    float v = row[c] * (cs ? cs[c] : 1.f);
    float q = fminf(fmaxf(rintf(v / scale), -128.f), 127.f);
    qw[(size_t)m * C_ + c] = (signed char)q;
    ssum += q;
  }
#pragma unroll
  for (int o = 16; o > 0; o >>= 1) ssum += __shfl_xor(ssum, o, 32);
  if ((t & 31) == 0) reds[t >> 5] = ssum;
  __syncthreads();
  if (t == 0){
    float s = 0.f;
    for (int i = 0; i < 8; ++i) s += reds[i];
    scw[m] = scale;
    rsw[m] = s;
  }
}

__global__ void k_quant_x(const float* x, const float* cs, const Scalars* sc, unsigned char* qx){
  size_t i = (size_t)blockIdx.x * 256 + threadIdx.x;
  float s = sc->scale[0], z = sc->zp[0];
  int c = (int)(i % C_);
  float q = fminf(fmaxf(rintf((x[i] / cs[c]) / s) + z, 0.f), 255.f);
  qx[i] = (unsigned char)q;
}

__global__ void k_quant_u8(const float* inp, const Scalars* sc, int slot, unsigned char* outq){
  size_t i = (size_t)blockIdx.x * 256 + threadIdx.x;
  float s = sc->scale[slot], z = sc->zp[slot];
  float q = fminf(fmaxf(rintf(inp[i] / s) + z, 0.f), 255.f);
  outq[i] = (unsigned char)q;
}

__global__ __launch_bounds__(256) void k_gemm_iu8(
    const unsigned char* __restrict__ A, const signed char* __restrict__ W,
    const float* __restrict__ bias, const float* __restrict__ scw,
    const float* __restrict__ rsw, Scalars* sc,
    int M, int slot_in, int mode, int slot_out,
    unsigned char* outq, float* outf)
{
  __shared__ alignas(16) unsigned char At[16 * C_];
  int rt = blockIdx.x, cg = blockIdx.y;
  int tid = threadIdx.x, lane = tid & 31, wv = tid >> 5;
  {
    unsigned ldsbase = (unsigned)(size_t)(void*)At;
    const unsigned char* src = A + (size_t)rt * 16 * C_;
#pragma unroll
    for (int i = 0; i < (16 * C_) / (256 * 16); ++i){
      unsigned off = (unsigned)(tid + i * 256) * 16u;
      unsigned lds = ldsbase + off;
      asm volatile("global_load_async_to_lds_b128 %0, %1, %2"
                   :: "v"(lds), "v"(off), "s"(src) : "memory");
    }
    asm volatile("s_wait_asynccnt 0" ::: "memory");
  }
  __syncthreads();
  int l16 = lane & 15, h = lane >> 4;
  int mcol = cg * 128 + wv * 16 + l16;
  __builtin_prefetch(W + (size_t)mcol * C_, 0, 0);
  v8i acc = {0,0,0,0,0,0,0,0};
  for (int kk = 0; kk < C_; kk += 64){
    v8i a, b;
#pragma unroll
    for (int v = 0; v < 8; ++v){
      int ka = (v >> 1) * 16 + (v & 1) * 4 + h * 8 + kk;
      a[v] = *(const int*)(At + l16 * C_ + ka);
      int kb = (v >> 1) * 16 + (v & 1) * 4 + h * 8 + kk;
      b[v] = *(const int*)(W + (size_t)mcol * C_ + kb);
    }
    acc = __builtin_amdgcn_wmma_i32_16x16x64_iu8(false, a, true, b, acc, false, false);
  }
  float s_in = sc->scale[slot_in], z_in = sc->zp[slot_in];
  float sw = scw[mcol] * s_in, corr = z_in * rsw[mcol], bb = bias[mcol];
  float so = 1.f, zo = 0.f;
  if (mode) { so = sc->scale[slot_out]; zo = sc->zp[slot_out]; }
  float mn = 3.4e38f, mx = -3.4e38f;
#pragma unroll
  for (int r = 0; r < 8; ++r){
    int row = rt * 16 + r + 8 * h;
    float val = ((float)acc[r] - corr) * sw + bb;
    if (mode == 0){
      mn = fminf(mn, val); mx = fmaxf(mx, val);
    } else if (mode == 1){
      float q = fminf(fmaxf(rintf(val / so) + zo, 0.f), 255.f);
      outq[(size_t)row * M + mcol] = (unsigned char)q;
    } else {
      float q = fminf(fmaxf(rintf(val / so) + zo, 0.f), 255.f);
      outf[(size_t)row * M + mcol] = (q - zo) * so;
    }
  }
  if (mode == 0) wave_mm_atomic(mn, mx, sc->mm[slot_out]);
}

__device__ __forceinline__ void attn_strip_compute(
    const unsigned char* __restrict__ qh, int b, int head, int rt,
    float s1, float z1, unsigned char* Qt, unsigned char* Kt,
    float* qsum, float* ksum, int lane, float a13[NT][8])
{
  int qoff = head * HD;
  int koff = C_ + head * HD;
#pragma unroll
  for (int s = 0; s < 8; ++s){
    int idx = lane + 32 * s;
    int r = idx >> 4, c4 = (idx & 15) * 4;
    int n = rt * 16 + r;
    unsigned val = 0u;
    if (n < N_) val = *(const unsigned*)(qh + (size_t)(b * N_ + n) * M3 + qoff + c4);
    *(unsigned*)(Qt + r * 64 + c4) = val;
  }
  for (int idx = lane; idx < NP * 16; idx += 32){
    int r = idx >> 4, c4 = (idx & 15) * 4;
    unsigned val = 0u;
    if (r < N_) val = *(const unsigned*)(qh + (size_t)(b * N_ + r) * M3 + koff + c4);
    *(unsigned*)(Kt + r * 64 + c4) = val;
  }
  __syncthreads();
  if (lane < 16){
    float s = 0.f;
    for (int d = 0; d < HD; ++d) s += (float)Qt[lane * 64 + d];
    qsum[lane] = s;
  }
  for (int r = lane; r < NP; r += 32){
    float s = 0.f;
    for (int d = 0; d < HD; ++d) s += (float)Kt[r * 64 + d];
    ksum[r] = s;
  }
  __syncthreads();
  int l16 = lane & 15, h = lane >> 4;
  v8i af;
#pragma unroll
  for (int v = 0; v < 8; ++v){
    int ka = (v >> 1) * 16 + (v & 1) * 4 + h * 8;
    af[v] = *(const int*)(Qt + l16 * 64 + ka);
  }
  float qs[8];
#pragma unroll
  for (int r = 0; r < 8; ++r) qs[r] = qsum[r + 8 * h];
  float c0 = s1 * s1 * 0.125f;
  float zz = 64.f * z1 * z1;
#pragma unroll
  for (int t = 0; t < NT; ++t){
    v8i bf;
#pragma unroll
    for (int v = 0; v < 8; ++v){
      int kb = (v >> 1) * 16 + (v & 1) * 4 + h * 8;
      bf[v] = *(const int*)(Kt + (t * 16 + l16) * 64 + kb);
    }
    v8i acc = {0,0,0,0,0,0,0,0};
    acc = __builtin_amdgcn_wmma_i32_16x16x64_iu8(false, af, false, bf, acc, false, false);
    float ks = ksum[t * 16 + l16];
#pragma unroll
    for (int r = 0; r < 8; ++r)
      a13[t][r] = ((float)acc[r] - z1 * (qs[r] + ks) + zz) * c0;
  }
}

__global__ __launch_bounds__(32) void k_attn_minmax(const unsigned char* __restrict__ qh, Scalars* sc){
  __shared__ alignas(16) unsigned char Qt[16 * 64];
  __shared__ alignas(16) unsigned char Kt[NP * 64];
  __shared__ float qsum[16], ksum[NP];
  int rt = blockIdx.x, head = blockIdx.y, b = blockIdx.z, lane = threadIdx.x;
  float a13[NT][8];
  attn_strip_compute(qh, b, head, rt, sc->scale[1], sc->zp[1], Qt, Kt, qsum, ksum, lane, a13);
  int l16 = lane & 15, h = lane >> 4;
  float mn = 3.4e38f, mx = -3.4e38f;
#pragma unroll
  for (int t = 0; t < NT; ++t){
    int j = t * 16 + l16;
#pragma unroll
    for (int r = 0; r < 8; ++r){
      int i = rt * 16 + r + 8 * h;
      if (i < N_ && j < N_){
        mn = fminf(mn, a13[t][r]);
        mx = fmaxf(mx, a13[t][r]);
      }
    }
  }
  wave_mm_atomic(mn, mx, sc->mm[2]);
}

__global__ __launch_bounds__(32) void k_attn_softmax_av(const unsigned char* __restrict__ qh,
                                                        Scalars* sc, float* __restrict__ outf){
  __shared__ alignas(16) unsigned char Qt[16 * 64];
  __shared__ alignas(16) unsigned char Kt[NP * 64];
  __shared__ float qsum[16], ksum[NP];
  __shared__ alignas(16) _Float16 strip[16 * NP2];
  __shared__ alignas(16) _Float16 vT[HD * NP2];
  int rt = blockIdx.x, head = blockIdx.y, b = blockIdx.z, lane = threadIdx.x;
  float s1 = sc->scale[1], z1 = sc->zp[1];
  float as = sc->scale[2], az = sc->zp[2];
  float a13[NT][8];
  attn_strip_compute(qh, b, head, rt, s1, z1, Qt, Kt, qsum, ksum, lane, a13);

  int voff = 2 * C_ + head * HD;
  for (int j = lane; j < NP2; j += 32){
    for (int d = 0; d < HD; ++d){
      float v = 0.f;
      if (j < N_) v = ((float)qh[(size_t)(b * N_ + j) * M3 + voff + d] - z1) * s1;
      vT[d * NP2 + j] = (_Float16)v;
    }
  }

  int l16 = lane & 15, h = lane >> 4;
#pragma unroll
  for (int t = 0; t < NT; ++t){
    int j = t * 16 + l16;
#pragma unroll
    for (int r = 0; r < 8; ++r){
      float q = fminf(fmaxf(rintf(a13[t][r] / as) + az, 0.f), 255.f);
      a13[t][r] = (j < N_) ? (q - az) : -1e30f;
    }
  }
  float pm[8];
#pragma unroll
  for (int r = 0; r < 8; ++r){
    float v = -1e30f;
#pragma unroll
    for (int t = 0; t < NT; ++t) v = fmaxf(v, a13[t][r]);
#pragma unroll
    for (int o = 1; o < 16; o <<= 1) v = fmaxf(v, __shfl_xor(v, o, 32));
    pm[r] = v;
  }
  float x0 = floorf(-0.6931f / as);
  float bI = floorf(0.96963238f / 0.35815147f / as);
  float cI = floorf(1.0f / (0.35815147f * as * as));
  float es[8];
#pragma unroll
  for (int r = 0; r < 8; ++r) es[r] = 0.f;
#pragma unroll
  for (int t = 0; t < NT; ++t){
    int j = t * 16 + l16;
#pragma unroll
    for (int r = 0; r < 8; ++r){
      float e = 0.f;
      if (j < N_){
        float xi = fmaxf(a13[t][r] - pm[r], 30.f * x0);
        float qf = floorf(xi / x0);
        float rr = xi - x0 * qf;
        float zp = rr * (rr + bI) + cI;
        e = fmaxf(floorf(zp * exp2f(30.f - qf)), 0.f);
      }
      a13[t][r] = e;
      es[r] += e;
    }
  }
#pragma unroll
  for (int r = 0; r < 8; ++r){
    float v = es[r];
#pragma unroll
    for (int o = 1; o < 16; o <<= 1) v += __shfl_xor(v, o, 32);
    es[r] = v;
  }
#pragma unroll
  for (int t = 0; t < NT; ++t){
#pragma unroll
    for (int r = 0; r < 8; ++r){
      float e = a13[t][r];
      float ratio = (e > 0.f) ? rintf(es[r] / fmaxf(e, 1.f)) : 2147483648.f;
      float rl = round_ln_f(fmaxf(ratio, 1.f));
      float w = (rl >= 16.f) ? 0.f : exp2f(-fminf(fmaxf(rl, 0.f), 15.f));
      strip[(r + 8 * h) * NP2 + t * 16 + l16] = (_Float16)w;
    }
  }
  for (int idx = lane; idx < 16 * (NP2 - NP); idx += 32)
    strip[(idx >> 4) * NP2 + NP + (idx & 15)] = (_Float16)0.f;
  __syncthreads();

  float mn = 3.4e38f, mx = -3.4e38f;
#pragma unroll
  for (int nt = 0; nt < 4; ++nt){
    v8f acc = {0.f,0.f,0.f,0.f,0.f,0.f,0.f,0.f};
    for (int ks = 0; ks < NP2; ks += 32){
      union { v16h v; v8i i; } au, bu;
#pragma unroll
      for (int v = 0; v < 8; ++v){
        int ka = (v >> 2) * 16 + (v & 3) * 2 + h * 8 + ks;
        au.i[v] = *(const int*)((const char*)strip + ((size_t)l16 * NP2 + ka) * 2);
        int kb = (v >> 2) * 16 + (v & 3) * 2 + h * 8 + ks;
        bu.i[v] = *(const int*)((const char*)vT + ((size_t)(nt * 16 + l16) * NP2 + kb) * 2);
      }
      acc = __builtin_amdgcn_wmma_f32_16x16x32_f16(false, au.v, false, bu.v, (short)0, acc, false, false);
    }
#pragma unroll
    for (int r = 0; r < 8; ++r){
      int i = rt * 16 + r + 8 * h;
      if (i < N_){
        float val = acc[r];
        outf[(size_t)(b * N_ + i) * C_ + head * HD + nt * 16 + l16] = val;
        mn = fminf(mn, val); mx = fmaxf(mx, val);
      }
    }
  }
  wave_mm_atomic(mn, mx, sc->mm[3]);
}

static constexpr size_t OFF_SCAL = 0;
static constexpr size_t OFF_GMAX = 1024;
static constexpr size_t OFF_WMAX = OFF_GMAX + 4096;
static constexpr size_t OFF_CS   = OFF_WMAX + 4096;
static constexpr size_t OFF_QW   = OFF_CS + 4096;
static constexpr size_t OFF_SW   = OFF_QW + (size_t)M3 * C_;
static constexpr size_t OFF_RW   = OFF_SW + (size_t)M3 * 4;
static constexpr size_t OFF_QPW  = OFF_RW + (size_t)M3 * 4;
static constexpr size_t OFF_SPW  = OFF_QPW + (size_t)C_ * C_;
static constexpr size_t OFF_RPW  = OFF_SPW + (size_t)C_ * 4;
static constexpr size_t OFF_QX   = ((OFF_RPW + (size_t)C_ * 4 + 255) / 256) * 256;
static constexpr size_t OFF_QH   = OFF_QX + (size_t)BN * C_;
static constexpr size_t OFF_AO   = OFF_QH + (size_t)BN * M3;

extern "C" void kernel_launch(void* const* d_in, const int* in_sizes, int n_in,
                              void* d_out, int out_size, void* d_ws, size_t ws_size,
                              hipStream_t stream) {
  (void)in_sizes; (void)n_in; (void)out_size; (void)ws_size;
  const float* x      = (const float*)d_in[0];
  const float* qkv_w  = (const float*)d_in[1];
  const float* qkv_b  = (const float*)d_in[2];
  const float* proj_w = (const float*)d_in[3];
  const float* proj_b = (const float*)d_in[4];
  float* out = (float*)d_out;

  char* ws = (char*)d_ws;
  Scalars* sc = (Scalars*)(ws + OFF_SCAL);
  float* gmax = (float*)(ws + OFF_GMAX);
  float* wmax = (float*)(ws + OFF_WMAX);
  float* cs   = (float*)(ws + OFF_CS);
  signed char*   qw  = (signed char*)(ws + OFF_QW);
  float* scw = (float*)(ws + OFF_SW);
  float* rsw = (float*)(ws + OFF_RW);
  signed char*   qpw = (signed char*)(ws + OFF_QPW);
  float* spw = (float*)(ws + OFF_SPW);
  float* rpw = (float*)(ws + OFF_RPW);
  unsigned char* qx  = (unsigned char*)(ws + OFF_QX);
  unsigned char* qh  = (unsigned char*)(ws + OFF_QH);
  float* ao  = (float*)(ws + OFF_AO);

  dim3 gGemmQKV(BN / 16, M3 / 128), gGemmPROJ(BN / 16, C_ / 128), gAttn(NT, H_, B_);
  const int elemBlocks = (BN * C_) / 256;

  k_init<<<3, 256, 0, stream>>>(sc, (unsigned*)gmax, (unsigned*)wmax);
  k_absmax_cols<<<BN / 64, 256, 0, stream>>>(x, (unsigned*)gmax, 64);
  k_absmax_cols<<<M3 / 64, 256, 0, stream>>>(qkv_w, (unsigned*)wmax, 64);
  k_cs<<<3, 256, 0, stream>>>(gmax, wmax, cs);
  k_xs_minmax<<<BN / 64, 256, 0, stream>>>(x, cs, sc);
  k_finalize<<<1, 1, 0, stream>>>(sc, 0);
  k_quant_w<<<M3, 256, 0, stream>>>(qkv_w, cs, qw, scw, rsw);
  k_quant_x<<<elemBlocks, 256, 0, stream>>>(x, cs, sc, qx);

  k_gemm_iu8<<<gGemmQKV, 256, 0, stream>>>(qx, qw, qkv_b, scw, rsw, sc, M3, 0, 0, 1, nullptr, nullptr);
  k_finalize<<<1, 1, 0, stream>>>(sc, 1);
  k_gemm_iu8<<<gGemmQKV, 256, 0, stream>>>(qx, qw, qkv_b, scw, rsw, sc, M3, 0, 1, 1, qh, nullptr);

  k_attn_minmax<<<gAttn, 32, 0, stream>>>(qh, sc);
  k_finalize<<<1, 1, 0, stream>>>(sc, 2);
  k_attn_softmax_av<<<gAttn, 32, 0, stream>>>(qh, sc, ao);
  k_finalize<<<1, 1, 0, stream>>>(sc, 3);
  k_quant_u8<<<elemBlocks, 256, 0, stream>>>(ao, sc, 3, qx);

  k_quant_w<<<C_, 256, 0, stream>>>(proj_w, nullptr, qpw, spw, rpw);
  k_gemm_iu8<<<gGemmPROJ, 256, 0, stream>>>(qx, qpw, proj_b, spw, rpw, sc, C_, 3, 0, 4, nullptr, nullptr);
  k_finalize<<<1, 1, 0, stream>>>(sc, 4);
  k_gemm_iu8<<<gGemmPROJ, 256, 0, stream>>>(qx, qpw, proj_b, spw, rpw, sc, C_, 3, 2, 4, nullptr, out);
}
